// ChannelQuadLayer_16432544875288
// MI455X (gfx1250) — matrix operand order repaired and checked by reading, not compiled, not run on hardware
//
#include <hip/hip_runtime.h>

typedef __attribute__((ext_vector_type(16))) __bf16 v16bf;
typedef __attribute__((ext_vector_type(8)))  float  v8f;
typedef __attribute__((ext_vector_type(4)))  float  v4f;

#define C_IN   64
#define OUTC   256
#define NPAIR  2080                  // C*(C+1)/2
#define KTOT   2144                  // 64 + 2080 = 67*32
#define KCH    67                    // K chunks of 32
#define MT     32                    // pixels per block (2 WMMA M-subtiles)
#define NTILES 16                    // 256 / 16

// d_ws layout
#define WB_ELEMS (NTILES * KCH * 32 * 16)   // 548864 bf16 fragment elements
#define WB_BYTES (WB_ELEMS * 2)             // 1097728 bytes
#define II_OFF   WB_BYTES
#define JJ_OFF   (WB_BYTES + NPAIR)

__device__ __forceinline__ __bf16 to_bf16(float f) {
    union { float f; unsigned u; } v; v.f = f;
    unsigned r = v.u + 0x7FFFu + ((v.u >> 16) & 1u);   // round-to-nearest-even
    unsigned short h = (unsigned short)(r >> 16);
    __bf16 out;
    __builtin_memcpy(&out, &h, 2);
    return out;
}

// pack two f32 -> packed bf16x2 (lo = a, hi = b), RNE
__device__ __forceinline__ unsigned pack_bf16x2(float a, float b) {
    union { float f; unsigned u; } x, y; x.f = a; y.f = b;
    unsigned ra = x.u + 0x7FFFu + ((x.u >> 16) & 1u);
    unsigned rb = y.u + 0x7FFFu + ((y.u >> 16) & 1u);
    return (ra >> 16) | (rb & 0xFFFF0000u);
}

// ---------------- init: upper-triangular pair tables ----------------
__global__ void cq_build_pairs(unsigned char* ii, unsigned char* jj) {
    for (int p = threadIdx.x; p < NPAIR; p += blockDim.x) {
        int i = 0, start = 0;
        while (p >= start + (C_IN - i)) { start += C_IN - i; ++i; }
        ii[p] = (unsigned char)i;
        jj[p] = (unsigned char)(i + (p - start));
    }
}

// ---------------- init: fc_w f32 -> bf16 B-fragment layout ----------------
// wB[((nt*KCH + kc)*32 + lane)*16 + e] = bf16(fc_w[nt*16 + (lane&15)][kc*32 + KB])
// 32x16 B layout used here (contiguous k): lane holds N = lane&15; lanes 0-15 K=e, lanes 16-31 K=16+e
__global__ void cq_convert_w(const float* __restrict__ fc_w, __bf16* __restrict__ wB) {
    int idx = blockIdx.x * blockDim.x + threadIdx.x;
    #pragma unroll
    for (int q = 0; q < 4; ++q) {
        int id = idx * 4 + q;
        if (id >= WB_ELEMS) return;
        int e    = id & 15;
        int lane = (id >> 4) & 31;
        int t    = id >> 9;
        int kc   = t % KCH;
        int nt   = t / KCH;
        int k    = kc * 32 + ((lane >> 4) * 16 + e);
        int o    = nt * 16 + (lane & 15);
        wB[id] = to_bf16(fc_w[o * KTOT + k]);
    }
}

// ---------------- main: quad-feature expansion + WMMA GEMM ----------------
__global__ __launch_bounds__(256)
void cq_main(const float* __restrict__ x, const __bf16* __restrict__ wB,
             const unsigned char* __restrict__ ii, const unsigned char* __restrict__ jj,
             float* __restrict__ out) {
    extern __shared__ char smem[];
    __bf16* yA = (__bf16*)smem;                          // [KCH][2 msub][32 lane][16 e] = 137216 B
    float*  xs = (float*)(smem + KCH * 2 * 32 * 16 * 2); // [64 ch][32 px] = 8192 B

    const int tid  = threadIdx.x;
    const int lane = tid & 31;
    const int wv   = tid >> 5;                         // 8 waves

    const int P0 = blockIdx.x * MT;                    // flat pixel base (b*4096 + h*64 + w)
    const int b  = P0 >> 12;
    const int hw = P0 & 4095;
    const int xOff = (b << 18) + hw;                   // b*64*4096 + hw

    // ---- phase 1: x tile -> LDS via float4 (2 x global_load_b128/thread) ----
    #pragma unroll
    for (int q = 0; q < 2; ++q) {
        int idx4 = q * 256 + tid;                      // 512 float4 = 64*32 floats
        int c = idx4 >> 3, m4 = (idx4 & 7) * 4;
        *(v4f*)&xs[c * 32 + m4] = *(const v4f*)&x[xOff + c * 4096 + m4];
    }
    __syncthreads();

    // ---- phase 2: build features in A-fragment layout, packed b32 stores ----
    // even/odd k share (kc, half, L) and occupy consecutive e -> one u32 store
    for (int t = tid; t < (KTOT / 2); t += 256) {
        const int k0 = 2 * t;
        int i0, j0, i1, j1;
        if (k0 < C_IN) { i0 = k0; j0 = -1; i1 = k0 + 1; j1 = -1; }
        else {
            int p = k0 - C_IN;
            i0 = ii[p];     j0 = jj[p];
            i1 = ii[p + 1]; j1 = jj[p + 1];
        }
        const int kc = k0 >> 5, kk = k0 & 31;
        const int half = kk >> 4;
        const int e    = kk & 15;                      // even
        unsigned* dst = (unsigned*)(yA + (size_t)kc * 1024 + (half << 4) * 16 + e);
        #pragma unroll
        for (int m = 0; m < MT; ++m) {
            float f0 = (j0 < 0) ? xs[i0 * 32 + m] : xs[i0 * 32 + m] * xs[j0 * 32 + m];
            float f1 = (j1 < 0) ? xs[i1 * 32 + m] : xs[i1 * 32 + m] * xs[j1 * 32 + m];
            int msub = m >> 4, mm = m & 15;
            // element offset ((msub*32 + mm)*16)/2 in u32 units; e already even
            dst[(msub * 32 + mm) * 8] = pack_bf16x2(f0, f1);
        }
    }
    __syncthreads();

    // ---- phase 3: WMMA main loop (each wave: 2 N-tiles x 2 M-subtiles) ----
    v8f acc00 = {0,0,0,0,0,0,0,0}, acc01 = acc00, acc10 = acc00, acc11 = acc00;

    const __bf16* ap  = yA + lane * 16;
    const __bf16* bp0 = wB + (size_t)wv       * KCH * 512 + lane * 16;
    const __bf16* bp1 = wB + (size_t)(wv + 8) * KCH * 512 + lane * 16;

    for (int kc = 0; kc < KCH; ++kc) {
        v16bf a0 = *(const v16bf*)(ap  + kc * 1024);
        v16bf a1 = *(const v16bf*)(ap  + kc * 1024 + 512);
        v16bf b0 = *(const v16bf*)(bp0 + kc * 512);
        v16bf b1 = *(const v16bf*)(bp1 + kc * 512);
        acc00 = __builtin_amdgcn_wmma_f32_16x16x32_bf16(false, a0, false, b0, (short)0, acc00, false, false);
        acc01 = __builtin_amdgcn_wmma_f32_16x16x32_bf16(false, a0, false, b1, (short)0, acc01, false, false);
        acc10 = __builtin_amdgcn_wmma_f32_16x16x32_bf16(false, a1, false, b0, (short)0, acc10, false, false);
        acc11 = __builtin_amdgcn_wmma_f32_16x16x32_bf16(false, a1, false, b1, (short)0, acc11, false, false);
    }

    // ---- store: D layout -> per-lane 8 consecutive floats along W (NT stores) ----
    const int Lhi = lane >> 4, Ln = lane & 15;
    const int o0 = wv * 16 + Ln, o1 = (wv + 8) * 16 + Ln;
    #pragma unroll
    for (int pick = 0; pick < 4; ++pick) {
        int mi = pick >> 1, ni = pick & 1;
        v8f a = (pick == 0) ? acc00 : (pick == 1) ? acc01 : (pick == 2) ? acc10 : acc11;
        int o = ni ? o1 : o0;
        float* p = out + (((b * OUTC + o) << 12) + hw + mi * 16 + Lhi * 8);
        __builtin_nontemporal_store(__builtin_shufflevector(a, a, 0, 1, 2, 3), (v4f*)p);
        __builtin_nontemporal_store(__builtin_shufflevector(a, a, 4, 5, 6, 7), (v4f*)(p + 4));
    }
}

extern "C" void kernel_launch(void* const* d_in, const int* in_sizes, int n_in,
                              void* d_out, int out_size, void* d_ws, size_t ws_size,
                              hipStream_t stream) {
    const float* x    = (const float*)d_in[0];
    const float* fc_w = (const float*)d_in[1];
    float* out        = (float*)d_out;

    __bf16* wB        = (__bf16*)d_ws;
    unsigned char* ii = (unsigned char*)d_ws + II_OFF;
    unsigned char* jj = (unsigned char*)d_ws + JJ_OFF;

    cq_build_pairs<<<1, 256, 0, stream>>>(ii, jj);
    cq_convert_w<<<(WB_ELEMS + 1023) / 1024, 256, 0, stream>>>(fc_w, wB);

    const int nBlocks = (8 * 64 * 64) / MT;            // 32768 pixels / 32 = 1024
    const size_t shmem = (size_t)KCH * 2 * 32 * 16 * 2 + 64 * MT * 4; // 145408 B
    cq_main<<<nBlocks, 256, shmem, stream>>>(x, wB, ii, jj, out);
}
